// GATTransformerLayer_17557826306414
// MI455X (gfx1250) — matrix operand order repaired and checked by reading, not compiled, not run on hardware
//
#include <hip/hip_runtime.h>
#include <hip/hip_bf16.h>

// ---------------------------------------------------------------------------
// GAT transformer layer for gfx1250 (MI455X)
//   - v_wmma_f32_16x16x32_bf16 for all GEMMs (fp32 accumulate)
//   - global_load_async_to_lds_b128 staging of A row-blocks (ASYNCcnt path)
//   - atomic scatter-add message passing with folded softmax normalization
// ---------------------------------------------------------------------------

typedef __attribute__((ext_vector_type(16))) __bf16 bf16x16;
typedef __attribute__((ext_vector_type(8)))  float  f32x8;

union AFragU { bf16x16 v; unsigned u[8]; };

#define D_EMB 128
#define D_FF  512

// ----- async stage of A row-block (16 x K bf16) into LDS --------------------
// Cooperative across the 256-thread workgroup; each lane issues b128 async
// copies (global -> LDS, no VGPR data), then waits ASYNCcnt and barriers.
template <int K>
__device__ __forceinline__ void stage_a_async(const __bf16* A, int row0,
                                              const __bf16* smem, int t) {
  constexpr int CPR = (K * 2) / 16;   // 16B chunks per row
  constexpr int TOT = 16 * CPR;       // chunks for 16 rows
  unsigned sbase = (unsigned)(size_t)smem;  // low 32 bits = LDS byte offset
#pragma unroll
  for (int i = t; i < TOT; i += 256) {
    int r = i / CPR, c = i % CPR;
    unsigned loff = sbase + (unsigned)(r * (K * 2) + c * 16);
    const __bf16* g = A + (size_t)(row0 + r) * K + c * 8;
    asm volatile("global_load_async_to_lds_b128 %0, %1, off"
                 :: "v"(loff), "v"(g) : "memory");
  }
  asm volatile("s_wait_asynccnt 0" ::: "memory");
  __syncthreads();
}

// ----- WMMA fragment loaders (layouts per CDNA5 ISA 7.12.2) -----------------

// A-matrix 16x32 bf16 fragment from the LDS-staged row-block ([16][K] row-major)
// lane 0-15 : M=lane, v0..3 K=2j,2j+1 ; v4..7 K=16+2(j-4),+1
// lane16-31 : M=lane-16, K shifted by +8 within each half.
__device__ __forceinline__ bf16x16 load_a_frag_lds(const __bf16* S, int K,
                                                   int k0, int lane) {
  int m  = lane & 15;
  int h2 = lane >> 4;
  const __bf16* base = S + m * K + k0;
  AFragU f;
#pragma unroll
  for (int j = 0; j < 8; ++j) {
    int kk = (j < 4) ? (8 * h2 + 2 * j) : (16 + 8 * h2 + 2 * (j - 4));
    f.u[j] = *(const unsigned*)(base + kk);  // two contiguous bf16 (ds_load)
  }
  return f.v;
}

// B-matrix 32x16 bf16 fragment from a TRANSPOSED weight buffer Bt[n*ldk + k].
// lane 0-15 : N=lane, v0..3 K=2j,2j+1 ; v4..7 K=16+2(j-4),+1 ; lane16-31 : N=lane-16, K +8
__device__ __forceinline__ bf16x16 load_b_frag(const __bf16* Bt, int ldk,
                                               int col0, int k0, int lane) {
  int n  = lane & 15;
  int h2 = lane >> 4;
  const __bf16* base = Bt + (size_t)(col0 + n) * ldk + k0;
  AFragU f;
#pragma unroll
  for (int j = 0; j < 8; ++j) f.u[j] = *(const unsigned*)(base + ((j < 4) ? (8 * h2 + 2 * j) : (16 + 8 * h2 + 2 * (j - 4))));
  return f.v;
}

__device__ __forceinline__ f32x8 wmma_bf16(bf16x16 a, bf16x16 b, f32x8 c) {
  return __builtin_amdgcn_wmma_f32_16x16x32_bf16(
      /*neg_a=*/false, a, /*neg_b=*/false, b,
      /*c_mod=*/(short)0, c, /*reuse_a=*/false, /*reuse_b=*/false);
}

// ----- small utility kernels ------------------------------------------------

__global__ __launch_bounds__(256) void zero_f32(float* p, int n) {
  int i = blockIdx.x * 256 + threadIdx.x;
  if (i < n) p[i] = 0.0f;
}

__global__ __launch_bounds__(256) void cvt_bf16(const float* s, __bf16* d, int n) {
  int i = blockIdx.x * 256 + threadIdx.x;
  if (i < n) d[i] = (__bf16)s[i];
}

// transpose-convert: d[c*rows + r] = (bf16)s[r*cols + c]
__global__ __launch_bounds__(256) void cvt_bf16_t(const float* s, __bf16* d,
                                                  int rows, int cols) {
  int i = blockIdx.x * 256 + threadIdx.x;
  if (i < rows * cols) {
    int r = i / cols, c = i % cols;
    d[(size_t)c * rows + r] = (__bf16)s[i];
  }
}

// ----- GEMM kernels ---------------------------------------------------------
// Workgroup = 8 waves = 16 output rows x 128 output cols; each wave owns one
// 16x16 tile. A row-block staged in LDS via async copies and shared by all
// waves; B (weights) read direct (L2-resident bf16, distinct cols per wave).

__global__ __launch_bounds__(256) void gemm_h_kernel(const __bf16* xb,
                                                     const __bf16* Wt,
                                                     float* h) {
  __shared__ __bf16 smem[16 * D_EMB];
  int wave = threadIdx.x >> 5, lane = threadIdx.x & 31;
  int row0 = blockIdx.x * 16;
  int col0 = wave * 16;
  stage_a_async<D_EMB>(xb, row0, smem, threadIdx.x);
  f32x8 acc = {};
#pragma unroll
  for (int kt = 0; kt < 4; ++kt) {
    bf16x16 a = load_a_frag_lds(smem, D_EMB, kt * 32, lane);
    bf16x16 b = load_b_frag(Wt, D_EMB, col0, kt * 32, lane);
    acc = wmma_bf16(a, b, acc);
  }
  int m0 = row0 + 8 * (lane >> 4);
  int n  = col0 + (lane & 15);
#pragma unroll
  for (int j = 0; j < 8; ++j) h[(size_t)(m0 + j) * D_EMB + n] = acc[j];
}

__global__ __launch_bounds__(256) void gemm_ffn1_kernel(const __bf16* vb,
                                                        const __bf16* W1t,
                                                        const float* b1,
                                                        __bf16* mid) {
  __shared__ __bf16 smem[16 * D_EMB];
  int wave = threadIdx.x >> 5, lane = threadIdx.x & 31;
  int row0 = blockIdx.x * 16;
  int col0 = blockIdx.y * 128 + wave * 16;
  stage_a_async<D_EMB>(vb, row0, smem, threadIdx.x);
  f32x8 acc = {};
#pragma unroll
  for (int kt = 0; kt < 4; ++kt) {
    bf16x16 a = load_a_frag_lds(smem, D_EMB, kt * 32, lane);
    bf16x16 b = load_b_frag(W1t, D_EMB, col0, kt * 32, lane);
    acc = wmma_bf16(a, b, acc);
  }
  int m0 = row0 + 8 * (lane >> 4);
  int n  = col0 + (lane & 15);
  float bias = b1[n];
#pragma unroll
  for (int j = 0; j < 8; ++j) {
    float r = acc[j] + bias;
    r = r > 0.0f ? r : 0.0f;  // relu
    mid[(size_t)(m0 + j) * D_FF + n] = (__bf16)r;
  }
}

__global__ __launch_bounds__(256) void gemm_ffn2_kernel(const __bf16* mid,
                                                        const __bf16* W2t,
                                                        const float* b2,
                                                        float* ffo) {
  __shared__ __bf16 smem[16 * D_FF];   // 16 KB
  int wave = threadIdx.x >> 5, lane = threadIdx.x & 31;
  int row0 = blockIdx.x * 16;
  int col0 = wave * 16;
  stage_a_async<D_FF>(mid, row0, smem, threadIdx.x);
  f32x8 acc = {};
#pragma unroll
  for (int kt = 0; kt < 16; ++kt) {
    __builtin_prefetch(W2t + (size_t)(col0 + (lane & 15)) * D_FF + kt * 32, 0, 1);
    bf16x16 a = load_a_frag_lds(smem, D_FF, kt * 32, lane);
    bf16x16 b = load_b_frag(W2t, D_FF, col0, kt * 32, lane);
    acc = wmma_bf16(a, b, acc);
  }
  int m0 = row0 + 8 * (lane >> 4);
  int n  = col0 + (lane & 15);
  float bias = b2[n];
#pragma unroll
  for (int j = 0; j < 8; ++j) ffo[(size_t)(m0 + j) * D_EMB + n] = acc[j] + bias;
}

// ----- attention scalars: a_src[n] = h[n,:].att_src ; a_dst likewise --------

__global__ __launch_bounds__(256) void att_kernel(const float* h,
                                                  const float* att_src,
                                                  const float* att_dst,
                                                  float* a_src, float* a_dst,
                                                  int N) {
  int wave = threadIdx.x >> 5, lane = threadIdx.x & 31;
  int r = blockIdx.x * 8 + wave;
  if (r >= N) return;
  float4 hv = ((const float4*)(h + (size_t)r * D_EMB))[lane];
  float4 as = ((const float4*)att_src)[lane];
  float4 ad = ((const float4*)att_dst)[lane];
  float ss = hv.x * as.x + hv.y * as.y + hv.z * as.z + hv.w * as.w;
  float sd = hv.x * ad.x + hv.y * ad.y + hv.z * ad.z + hv.w * ad.w;
#pragma unroll
  for (int off = 16; off > 0; off >>= 1) {
    ss += __shfl_xor(ss, off);
    sd += __shfl_xor(sd, off);
  }
  if (lane == 0) { a_src[r] = ss; a_dst[r] = sd; }
}

// ----- edge passes ----------------------------------------------------------

__device__ __forceinline__ unsigned enc_f32(float f) {
  unsigned u = __float_as_uint(f);
  return (u & 0x80000000u) ? ~u : (u | 0x80000000u);
}
__device__ __forceinline__ float dec_f32(unsigned e) {
  return (e & 0x80000000u) ? __uint_as_float(e & 0x7FFFFFFFu)
                           : __uint_as_float(~e);
}

// pass 1: segment max of leaky-relu'd attention logit (thread per edge)
__global__ __launch_bounds__(256) void edge_pass1(const int* ei,
                                                  const float* a_src,
                                                  const float* a_dst,
                                                  unsigned* amax_u,
                                                  int E, int M) {
  int e = blockIdx.x * 256 + threadIdx.x;
  if (e >= M) return;
  int s, d;
  if (e < E) { s = ei[e]; d = ei[E + e]; } else { s = d = e - E; }
  float a = a_src[s] + a_dst[d];
  a = a > 0.0f ? a : 0.2f * a;
  atomicMax(amax_u + d, enc_f32(a));
}

// pass 2: wave per edge; unnormalized exp-weighted message scatter + denom
__global__ __launch_bounds__(256) void edge_pass2(const int* ei,
                                                  const float* a_src,
                                                  const float* a_dst,
                                                  const unsigned* amax_u,
                                                  const float* h,
                                                  float* denom, float* agg,
                                                  int E, int M) {
  int wave = threadIdx.x >> 5, lane = threadIdx.x & 31;
  int e = blockIdx.x * 8 + wave;
  if (e >= M) return;
  int s, d;
  if (e < E) { s = ei[e]; d = ei[E + e]; } else { s = d = e - E; }
  float a = a_src[s] + a_dst[d];
  a = a > 0.0f ? a : 0.2f * a;
  float ex = __expf(a - dec_f32(amax_u[d]));
  if (lane == 0) atomicAdd(denom + d, ex);
  float4 hv = ((const float4*)(h + (size_t)s * D_EMB))[lane];
  float* dst = agg + (size_t)d * D_EMB + lane * 4;
  atomicAdd(dst + 0, hv.x * ex);
  atomicAdd(dst + 1, hv.y * ex);
  atomicAdd(dst + 2, hv.z * ex);
  atomicAdd(dst + 3, hv.w * ex);
}

// ----- layernorm kernels (wave per row, 4 cols/lane) ------------------------

__global__ __launch_bounds__(256) void ln1_kernel(const float* agg,
                                                  const float* denom,
                                                  const float* gbias,
                                                  const float* x,
                                                  const float* g, const float* b,
                                                  float* v, __bf16* vb, int N) {
  int wave = threadIdx.x >> 5, lane = threadIdx.x & 31;
  int r = blockIdx.x * 8 + wave;
  if (r >= N) return;
  float inv = 1.0f / (denom[r] + 1e-16f);
  float4 a4 = ((const float4*)(agg + (size_t)r * D_EMB))[lane];
  float4 x4 = ((const float4*)(x + (size_t)r * D_EMB))[lane];
  float4 gb = ((const float4*)gbias)[lane];
  float val[4] = { a4.x * inv + gb.x + x4.x, a4.y * inv + gb.y + x4.y,
                   a4.z * inv + gb.z + x4.z, a4.w * inv + gb.w + x4.w };
  float s = val[0] + val[1] + val[2] + val[3];
#pragma unroll
  for (int off = 16; off > 0; off >>= 1) s += __shfl_xor(s, off);
  float mean = s * (1.0f / 128.0f);
  float vs = 0.0f;
#pragma unroll
  for (int i = 0; i < 4; ++i) { float t = val[i] - mean; vs += t * t; }
#pragma unroll
  for (int off = 16; off > 0; off >>= 1) vs += __shfl_xor(vs, off);
  float rstd = rsqrtf(vs * (1.0f / 128.0f) + 1e-5f);
  float4 g4 = ((const float4*)g)[lane];
  float4 b4 = ((const float4*)b)[lane];
  float o0 = (val[0] - mean) * rstd * g4.x + b4.x;
  float o1 = (val[1] - mean) * rstd * g4.y + b4.y;
  float o2 = (val[2] - mean) * rstd * g4.z + b4.z;
  float o3 = (val[3] - mean) * rstd * g4.w + b4.w;
  float4* vp = (float4*)(v + (size_t)r * D_EMB);
  vp[lane] = make_float4(o0, o1, o2, o3);
  __bf16* q = vb + (size_t)r * D_EMB + lane * 4;
  q[0] = (__bf16)o0; q[1] = (__bf16)o1; q[2] = (__bf16)o2; q[3] = (__bf16)o3;
}

__global__ __launch_bounds__(256) void ln2_kernel(const float* v,
                                                  const float* ffo,
                                                  const float* g, const float* b,
                                                  float* out, int N) {
  int wave = threadIdx.x >> 5, lane = threadIdx.x & 31;
  int r = blockIdx.x * 8 + wave;
  if (r >= N) return;
  float4 v4 = ((const float4*)(v + (size_t)r * D_EMB))[lane];
  float4 f4 = ((const float4*)(ffo + (size_t)r * D_EMB))[lane];
  float val[4] = { v4.x + f4.x, v4.y + f4.y, v4.z + f4.z, v4.w + f4.w };
  float s = val[0] + val[1] + val[2] + val[3];
#pragma unroll
  for (int off = 16; off > 0; off >>= 1) s += __shfl_xor(s, off);
  float mean = s * (1.0f / 128.0f);
  float vs = 0.0f;
#pragma unroll
  for (int i = 0; i < 4; ++i) { float t = val[i] - mean; vs += t * t; }
#pragma unroll
  for (int off = 16; off > 0; off >>= 1) vs += __shfl_xor(vs, off);
  float rstd = rsqrtf(vs * (1.0f / 128.0f) + 1e-5f);
  float4 g4 = ((const float4*)g)[lane];
  float4 b4 = ((const float4*)b)[lane];
  float4* op = (float4*)(out + (size_t)r * D_EMB);
  op[lane] = make_float4((val[0] - mean) * rstd * g4.x + b4.x,
                         (val[1] - mean) * rstd * g4.y + b4.y,
                         (val[2] - mean) * rstd * g4.z + b4.z,
                         (val[3] - mean) * rstd * g4.w + b4.w);
}

// ----- host-side orchestration ----------------------------------------------

extern "C" void kernel_launch(void* const* d_in, const int* in_sizes, int n_in,
                              void* d_out, int out_size, void* d_ws, size_t ws_size,
                              hipStream_t stream) {
  const float* x       = (const float*)d_in[0];
  const int*   ei      = (const int*)d_in[1];   // [2,E] int32
  const float* W       = (const float*)d_in[3];
  const float* att_src = (const float*)d_in[4];
  const float* att_dst = (const float*)d_in[5];
  const float* gbias   = (const float*)d_in[6];
  const float* W1      = (const float*)d_in[7];
  const float* b1      = (const float*)d_in[8];
  const float* W2      = (const float*)d_in[9];
  const float* b2      = (const float*)d_in[10];
  const float* ln1g    = (const float*)d_in[11];
  const float* ln1b    = (const float*)d_in[12];
  const float* ln2g    = (const float*)d_in[13];
  const float* ln2b    = (const float*)d_in[14];

  const int N = in_sizes[0] / D_EMB;   // 50000 (divisible by 16)
  const int E = in_sizes[1] / 2;       // 800000
  const int M = E + N;                 // edges incl. self loops
  float* out = (float*)d_out;          // doubles as message aggregate buffer

  // workspace carve-out (256B aligned slices)
  char* w = (char*)d_ws;
  auto alloc = [&](size_t bytes) -> char* {
    char* p = w;
    w += (bytes + 255) & ~(size_t)255;
    return p;
  };
  __bf16*   xb    = (__bf16*)alloc((size_t)N * D_EMB * 2);
  __bf16*   Wt    = (__bf16*)alloc((size_t)D_EMB * D_EMB * 2);
  __bf16*   W1t   = (__bf16*)alloc((size_t)D_EMB * D_FF * 2);
  __bf16*   W2t   = (__bf16*)alloc((size_t)D_FF * D_EMB * 2);
  float*    h     = (float*)alloc((size_t)N * D_EMB * 4);   // reused as ffo later
  float*    a_src = (float*)alloc((size_t)N * 4);
  float*    a_dst = (float*)alloc((size_t)N * 4);
  unsigned* amax  = (unsigned*)alloc((size_t)N * 4);
  float*    denom = (float*)alloc((size_t)N * 4);
  float*    v     = (float*)alloc((size_t)N * D_EMB * 4);
  __bf16*   vb    = (__bf16*)alloc((size_t)N * D_EMB * 2);
  __bf16*   mid   = (__bf16*)alloc((size_t)N * D_FF * 2);

  const int TB = 256;
  auto blocks = [](long long n, int t) { return (unsigned)((n + t - 1) / t); };

  // 0) zero accumulators (d_out is the scatter target; harness poisons it)
  zero_f32<<<blocks((long long)N * D_EMB, TB), TB, 0, stream>>>(out, N * D_EMB);
  zero_f32<<<blocks(N, TB), TB, 0, stream>>>(denom, N);
  zero_f32<<<blocks(N, TB), TB, 0, stream>>>((float*)amax, N);

  // 1) precision conversion (weights transposed for WMMA B-fragment loads)
  cvt_bf16<<<blocks((long long)N * D_EMB, TB), TB, 0, stream>>>(x, xb, N * D_EMB);
  cvt_bf16_t<<<blocks(D_EMB * D_EMB, TB), TB, 0, stream>>>(W, Wt, D_EMB, D_EMB);
  cvt_bf16_t<<<blocks(D_EMB * D_FF, TB), TB, 0, stream>>>(W1, W1t, D_EMB, D_FF);
  cvt_bf16_t<<<blocks(D_FF * D_EMB, TB), TB, 0, stream>>>(W2, W2t, D_FF, D_EMB);

  // 2) h = x @ W   (WMMA bf16 -> f32, async-LDS staged A)
  gemm_h_kernel<<<N / 16, TB, 0, stream>>>(xb, Wt, h);

  // 3) attention scalars
  att_kernel<<<blocks(N, 8), TB, 0, stream>>>(h, att_src, att_dst, a_src, a_dst, N);

  // 4) segment max, then exp-weighted scatter (normalization folded into LN1)
  edge_pass1<<<blocks(M, TB), TB, 0, stream>>>(ei, a_src, a_dst, amax, E, M);
  edge_pass2<<<blocks(M, 8), TB, 0, stream>>>(ei, a_src, a_dst, amax, h,
                                              denom, out, E, M);

  // 5) LN1( agg/denom + bias + x ) -> v (f32) and vb (bf16)
  ln1_kernel<<<blocks(N, 8), TB, 0, stream>>>(out, denom, gbias, x,
                                              ln1g, ln1b, v, vb, N);

  // 6) FFN: mid = relu(v@W1+b1) ; ffo = mid@W2+b2  (ffo reuses h buffer)
  gemm_ffn1_kernel<<<dim3(N / 16, D_FF / 128), TB, 0, stream>>>(vb, W1t, b1, mid);
  gemm_ffn2_kernel<<<N / 16, TB, 0, stream>>>(mid, W2t, b2, h);

  // 7) LN2( v + ffo ) -> d_out
  ln2_kernel<<<blocks(N, 8), TB, 0, stream>>>(v, h, ln2g, ln2b, out, N);
}
